// _Int8Linear_1795296330415
// MI455X (gfx1250) — 8-bit matrix operand order repaired and checked by reading, compiled, not run on hardware
//
#include <hip/hip_runtime.h>
#include <stdint.h>


typedef __attribute__((ext_vector_type(8))) int v8i;

#define D_MODEL   1024
#define VOCAB     128000
#define M_TOTAL   1024

#define BM        128
#define BN        128
#define KT        64
#define NCHUNK    (D_MODEL / KT)
#define LSTRIDE   80
#define NTHREADS  256

__device__ __forceinline__ void async_load_b128_s(uint32_t lds_off, uint32_t voff,
                                                  const void* sbase)
{
    asm volatile("global_load_async_to_lds_b128 %0, %1, %2"
                 :: "v"(lds_off), "v"(voff), "s"(sbase)
                 : "memory");
}

__device__ __forceinline__ void wait_async0()
{
#if __has_builtin(__builtin_amdgcn_s_wait_asynccnt)
    __builtin_amdgcn_s_wait_asynccnt(0);
#else
    asm volatile("s_wait_asynccnt 0x0" ::: "memory");
#endif
}

__global__ __launch_bounds__(NTHREADS) void quant_rows_kernel(
    const float* __restrict__ x, int8_t* __restrict__ xq, float* __restrict__ rs)
{
    __shared__ float red[NTHREADS];
    const int row = blockIdx.x;
    const int t   = threadIdx.x;

    const float4 v = ((const float4*)(x + (size_t)row * D_MODEL))[t];
    float m = fmaxf(fmaxf(fabsf(v.x), fabsf(v.y)), fmaxf(fabsf(v.z), fabsf(v.w)));
    red[t] = m;
    __syncthreads();
#pragma unroll
    for (int s = NTHREADS / 2; s > 0; s >>= 1) {
        if (t < s) red[t] = fmaxf(red[t], red[t + s]);
        __syncthreads();
    }
    const float absmax = fmaxf(red[0], 1e-8f);
    const float scale  = 127.0f / absmax;
    if (t == 0) rs[row] = absmax * (1.0f / 127.0f);

    float q0 = fminf(fmaxf(rintf(v.x * scale), -127.0f), 127.0f);
    float q1 = fminf(fmaxf(rintf(v.y * scale), -127.0f), 127.0f);
    float q2 = fminf(fmaxf(rintf(v.z * scale), -127.0f), 127.0f);
    float q3 = fminf(fmaxf(rintf(v.w * scale), -127.0f), 127.0f);
    int packed = ((int)q0 & 0xff) | (((int)q1 & 0xff) << 8) |
                 (((int)q2 & 0xff) << 16) | (((int)q3 & 0xff) << 24);
    ((int*)xq)[(size_t)row * (D_MODEL / 4) + t] = packed;
}

__global__ __launch_bounds__(NTHREADS) void gemm_i8_kernel(
    const int8_t* __restrict__ xq,
    const float*  __restrict__ rs,
    const int8_t* __restrict__ w,
    const float*  __restrict__ wscale,
    const float*  __restrict__ bias,
    float*        __restrict__ out)
{
    __shared__ char lA[2][BM * LSTRIDE];
    __shared__ char lB[2][BN * LSTRIDE];

    const int nBlock = blockIdx.x * BN;
    const int mBlock = blockIdx.y * BM;
    const int t      = threadIdx.x;
    const int lane   = t & 31;
    const int wv     = t >> 5;
    const int waveM  = (wv & 1) * 64;
    const int waveN  = (wv >> 1) * 32;
    const int ml     = lane & 15;
    const int half   = lane >> 4;

    const int8_t* baseA = xq + (size_t)mBlock * D_MODEL;
    const int8_t* baseB = w  + (size_t)nBlock * D_MODEL;

    const int c0   = t;
    const int c1   = t + NTHREADS;
    const int row0 = c0 >> 2, col0 = (c0 & 3) * 16;
    const int row1 = c1 >> 2, col1 = (c1 & 3) * 16;
    const uint32_t voff0 = (uint32_t)(row0 * D_MODEL + col0);
    const uint32_t voff1 = (uint32_t)(row1 * D_MODEL + col1);
    const uint32_t ldsOff0 = (uint32_t)(row0 * LSTRIDE + col0);
    const uint32_t ldsOff1 = (uint32_t)(row1 * LSTRIDE + col1);
    const uint32_t ldsBaseA = (uint32_t)(uintptr_t)&lA[0][0];
    const uint32_t ldsBaseB = (uint32_t)(uintptr_t)&lB[0][0];

    auto issue_chunk = [&](int buf, int k0) {
        const uint32_t ab = ldsBaseA + (uint32_t)(buf * BM * LSTRIDE);
        const uint32_t bb = ldsBaseB + (uint32_t)(buf * BN * LSTRIDE);
        async_load_b128_s(ab + ldsOff0, voff0 + (uint32_t)k0, baseA);
        async_load_b128_s(bb + ldsOff0, voff0 + (uint32_t)k0, baseB);
        async_load_b128_s(ab + ldsOff1, voff1 + (uint32_t)k0, baseA);
        async_load_b128_s(bb + ldsOff1, voff1 + (uint32_t)k0, baseB);
    };

    v8i acc[4][2];
#pragma unroll
    for (int mt = 0; mt < 4; ++mt)
#pragma unroll
        for (int nt = 0; nt < 2; ++nt)
#pragma unroll
            for (int r = 0; r < 8; ++r) acc[mt][nt][r] = 0;

    issue_chunk(0, 0);

    for (int kc = 0; kc < NCHUNK; ++kc) {
        const int cur = kc & 1;

        wait_async0();
        __syncthreads();

        if (kc + 1 < NCHUNK)
            issue_chunk(cur ^ 1, (kc + 1) * KT);

        v8i afr[4];
#pragma unroll
        for (int mt = 0; mt < 4; ++mt) {
            const char* ap = &lA[cur][(waveM + mt * 16 + ml) * LSTRIDE + half * 8];
            int2 p0 = *(const int2*)(ap);
            int2 p1 = *(const int2*)(ap + 16);
            int2 p2 = *(const int2*)(ap + 32);
            int2 p3 = *(const int2*)(ap + 48);
            afr[mt][0] = p0.x; afr[mt][1] = p0.y;
            afr[mt][2] = p1.x; afr[mt][3] = p1.y;
            afr[mt][4] = p2.x; afr[mt][5] = p2.y;
            afr[mt][6] = p3.x; afr[mt][7] = p3.y;
        }

#pragma unroll
        for (int nt = 0; nt < 2; ++nt) {
            const char* bp = &lB[cur][(waveN + nt * 16 + ml) * LSTRIDE + half * 8];
            int2 q0 = *(const int2*)(bp);
            int2 q1 = *(const int2*)(bp + 16);
            int2 q2 = *(const int2*)(bp + 32);
            int2 q3 = *(const int2*)(bp + 48);
            v8i bfr;
            bfr[0] = q0.x; bfr[1] = q0.y; bfr[2] = q1.x; bfr[3] = q1.y;
            bfr[4] = q2.x; bfr[5] = q2.y; bfr[6] = q3.x; bfr[7] = q3.y;
#pragma unroll
            for (int mt = 0; mt < 4; ++mt) {
                acc[mt][nt] = __builtin_amdgcn_wmma_i32_16x16x64_iu8(
                    true, afr[mt], true, bfr, acc[mt][nt], false, false);
            }
        }
    }

#pragma unroll
    for (int nt = 0; nt < 2; ++nt) {
        const int n = nBlock + waveN + nt * 16 + ml;
        const float inv_ws = 1.0f / wscale[n];
        const float bv     = bias[n];
#pragma unroll
        for (int mt = 0; mt < 4; ++mt) {
#pragma unroll
            for (int r = 0; r < 8; ++r) {
                const int m = mBlock + waveM + mt * 16 + r + half * 8;
                const float val = (float)acc[mt][nt][r] * rs[m] * inv_ws + bv;
                out[(size_t)m * VOCAB + n] = val;
            }
        }
    }
}

extern "C" void kernel_launch(void* const* d_in, const int* in_sizes, int n_in,
                              void* d_out, int out_size, void* d_ws, size_t ws_size,
                              hipStream_t stream)
{
    const float*  x      = (const float*)d_in[0];
    const int8_t* w_i8   = (const int8_t*)d_in[1];
    const float*  wscale = (const float*)d_in[2];
    const float*  bias   = (const float*)d_in[3];
    float*        out    = (float*)d_out;

    int8_t* xq = (int8_t*)d_ws;
    float*  rs = (float*)((char*)d_ws + (size_t)M_TOTAL * D_MODEL);

    quant_rows_kernel<<<M_TOTAL, NTHREADS, 0, stream>>>(x, xq, rs);

    dim3 grid(VOCAB / BN, M_TOTAL / BM);
    gemm_i8_kernel<<<grid, NTHREADS, 0, stream>>>(xq, rs, w_i8, wscale, bias, out);
}
